// NC_24927990186534
// MI455X (gfx1250) — matrix operand order repaired and checked by reading, not compiled, not run on hardware
//
#include <hip/hip_runtime.h>
#include <math.h>

// ---------------------------------------------------------------------------
// Types for CDNA5 WMMA (wave32): v_wmma_f32_16x16x32_bf16
// ---------------------------------------------------------------------------
typedef __bf16 bf16_t;
typedef __attribute__((ext_vector_type(16))) __bf16 v16bf;
typedef __attribute__((ext_vector_type(8)))  __bf16 v8bf;
typedef __attribute__((ext_vector_type(8)))  float  v8f;

__device__ __forceinline__ v8f v8f_zero() {
    v8f z;
#pragma unroll
    for (int i = 0; i < 8; ++i) z[i] = 0.0f;
    return z;
}
__device__ __forceinline__ v16bf v16bf_zero() {
    v16bf z;
#pragma unroll
    for (int i = 0; i < 16; ++i) z[i] = (__bf16)0.0f;
    return z;
}

// A-matrix (16x32, MxK, bf16) per-lane fragment:
//   lane = half*16 + m ; VGPR0-3 hold K = half*8 + 0..7, VGPR4-7 hold K = 16 + half*8 + 0..7
// Row-major [row][K] storage -> two contiguous 16B loads.
__device__ __forceinline__ v16bf load_a16(const bf16_t* __restrict__ row, int k0, int half) {
    const v8bf lo = *(const v8bf*)(row + k0 + half * 8);
    const v8bf hi = *(const v8bf*)(row + k0 + 16 + half * 8);
    return __builtin_shufflevector(lo, hi, 0,1,2,3,4,5,6,7,8,9,10,11,12,13,14,15);
}
// B-matrix (32x16, KxN, bf16) per-lane fragment:
//   lane = half*16 + n ; VGPR0-3 hold K = half*8 + 0..7, VGPR4-7 hold K = 16 + half*8 + 0..7
// Row-major [n][K] storage -> two contiguous 16B loads.
__device__ __forceinline__ v16bf load_b16(const bf16_t* __restrict__ row, int k0, int half) {
    const v8bf lo = *(const v8bf*)(row + k0 + half * 8);
    const v8bf hi = *(const v8bf*)(row + k0 + 16 + half * 8);
    return __builtin_shufflevector(lo, hi, 0,1,2,3,4,5,6,7,8,9,10,11,12,13,14,15);
}
__device__ __forceinline__ v8f wmma_bf16(v16bf a, v16bf b, v8f c) {
    // (neg_a, A, neg_b, B, c_mod, C, reuse_a, reuse_b)
    return __builtin_amdgcn_wmma_f32_16x16x32_bf16(false, a, false, b, (short)0, c, false, false);
}

// ---------------------------------------------------------------------------
// Problem constants
// ---------------------------------------------------------------------------
#define LVL 9
#define BB  4
#define CC  512
#define HW  1024   // 32*32

// Workspace offsets (bytes). Overlays are safe given the launch order below.
#define QN_OFF    ((size_t)0)                      // bf16 [L,B,1024,512]  37.75MB
#define SN_OFF    ((size_t)37748736)               // bf16 [L,B,1024,512]  37.75MB
#define CORR_OFF  ((size_t)75497472)               // f32  [B,1024,1024]   16.78MB
#define FSB_OFF   ((size_t)92274688)               // bf16 [B,512,1024]     4.19MB
// overlays in QN region (dead after k_corr):
#define ATTN_OFF  ((size_t)0)                      // bf16 [B,1024,1024]    8.39MB
#define ATTWS_OFF ((size_t)8388608)                // f32  [B,1024,512]     8.39MB
// overlays in SN region (dead after k_corr):
#define COMP_OFF  (SN_OFF)                         // bf16 [B,1024,1024]    8.39MB
#define W1T_OFF   (SN_OFF + 8388608)               // bf16 [9,256,1024]     4.72MB
#define X1_OFF    (SN_OFF + 13107200)              // bf16 [B,1024,256]     2.10MB
#define W2T_OFF   (SN_OFF + 15204352)              // bf16 [9,256,256]      1.18MB
#define X2_OFF    (SN_OFF + 16384000)              // f32  [B,900,256]      3.69MB
#define POOL_OFF  (SN_OFF + 20070400)              // f32  [B,100,256]      0.41MB

// ---------------------------------------------------------------------------
// K1: [C=512][P=1024] f32 -> [P][C] bf16 transpose, optional channel L2-norm.
//     16-row p-tiles staged through LDS (pad 513 to dodge bank conflicts).
// ---------------------------------------------------------------------------
__global__ void k_transpose_norm(const float* __restrict__ in, bf16_t* __restrict__ out,
                                 int out_row_stride, int out_col_off, int do_norm) {
    __shared__ float tile[16][513];
    __shared__ float red[16][16];
    __shared__ float invn[16];
    const int outer = blockIdx.x;
    const int p0 = blockIdx.y * 16;
    const int tid = threadIdx.x;          // 256 threads
    const int px = tid & 15;
    const int cy = tid >> 4;              // 0..15
    const float* src = in + (size_t)outer * CC * HW;
    float ssq = 0.f;
#pragma unroll 4
    for (int i = 0; i < 32; ++i) {
        int c = cy + i * 16;
        float v = src[(size_t)c * HW + p0 + px];
        tile[px][c] = v;
        ssq += v * v;
    }
    red[cy][px] = ssq;
    __syncthreads();
    if (tid < 16) {
        float s = 0.f;
#pragma unroll
        for (int j = 0; j < 16; ++j) s += red[j][tid];
        invn[tid] = 1.0f / fmaxf(sqrtf(s), 1e-12f);
    }
    __syncthreads();
#pragma unroll 4
    for (int i = 0; i < 32; ++i) {
        int idx = tid + i * 256;          // 0..8191
        int p = idx >> 9;
        int c = idx & 511;
        float v = tile[p][c];
        if (do_norm) v *= invn[p];
        out[(size_t)(p0 + p) * out_row_stride + out_col_off + c] = (bf16_t)v;
    }
}

// f32 -> bf16 elementwise (keeps layout)
__global__ void k_cvt_bf16(const float* __restrict__ in, bf16_t* __restrict__ out, int n) {
    int idx = blockIdx.x * 256 + threadIdx.x;
    if (idx < n) out[idx] = (bf16_t)in[idx];
}

// Weight transform: w[CO][CI][3][3] f32 -> out[tap][CO][CI] bf16 (CI contiguous)
__global__ void k_wxform(const float* __restrict__ w, bf16_t* __restrict__ out, int CO, int CI) {
    int idx = blockIdx.x * 256 + threadIdx.x;
    int total = CO * CI * 9;
    if (idx >= total) return;
    int ci = idx % CI;
    int rest = idx / CI;
    int co = rest % CO;
    int t = rest / CO;
    out[idx] = (bf16_t)w[((size_t)co * CI + ci) * 9 + t];
}

// ---------------------------------------------------------------------------
// K2: correlation. One wave computes a 32x32 tile of mean_l relu(Q_l S_l^T).
//     grid = (s_tiles=32, q_tiles=32, B)
// ---------------------------------------------------------------------------
__global__ void k_corr(const bf16_t* __restrict__ qn, const bf16_t* __restrict__ sn,
                       float* __restrict__ corr) {
    const int s0 = blockIdx.x * 32, q0 = blockIdx.y * 32, b = blockIdx.z;
    const int lane = threadIdx.x, m = lane & 15, half = lane >> 4;
    v8f m00 = v8f_zero(), m01 = v8f_zero(), m10 = v8f_zero(), m11 = v8f_zero();
    for (int l = 0; l < LVL; ++l) {
        const bf16_t* Q = qn + (size_t)(l * BB + b) * HW * CC;
        const bf16_t* S = sn + (size_t)(l * BB + b) * HW * CC;
        const bf16_t* a0r = Q + (size_t)(q0 + m) * CC;
        const bf16_t* a1r = Q + (size_t)(q0 + 16 + m) * CC;
        const bf16_t* b0r = S + (size_t)(s0 + m) * CC;
        const bf16_t* b1r = S + (size_t)(s0 + 16 + m) * CC;
        v8f c00 = v8f_zero(), c01 = v8f_zero(), c10 = v8f_zero(), c11 = v8f_zero();
#pragma unroll 4
        for (int kc = 0; kc < 16; ++kc) {
            const int k0 = kc * 32;
            v16bf a0 = load_a16(a0r, k0, half);
            v16bf a1 = load_a16(a1r, k0, half);
            v16bf bm0 = load_b16(b0r, k0, half);
            v16bf bm1 = load_b16(b1r, k0, half);
            c00 = wmma_bf16(a0, bm0, c00);
            c01 = wmma_bf16(a0, bm1, c01);
            c10 = wmma_bf16(a1, bm0, c10);
            c11 = wmma_bf16(a1, bm1, c11);
        }
#pragma unroll
        for (int r = 0; r < 8; ++r) {
            m00[r] += fmaxf(c00[r], 0.f);
            m01[r] += fmaxf(c01[r], 0.f);
            m10[r] += fmaxf(c10[r], 0.f);
            m11[r] += fmaxf(c11[r], 0.f);
        }
    }
    const float inv9 = 1.0f / 9.0f;
    float* out = corr + (size_t)b * HW * HW;
#pragma unroll
    for (int r = 0; r < 8; ++r) {
        int row0 = q0 + r + 8 * half, row1 = row0 + 16;
        out[(size_t)row0 * HW + s0 + m]      = m00[r] * inv9;
        out[(size_t)row0 * HW + s0 + 16 + m] = m01[r] * inv9;
        out[(size_t)row1 * HW + s0 + m]      = m10[r] * inv9;
        out[(size_t)row1 * HW + s0 + 16 + m] = m11[r] * inv9;
    }
}

// ---------------------------------------------------------------------------
// K3: softmax(corr*20) over s; one 256-thread block per (b,q) row; bf16 out.
// ---------------------------------------------------------------------------
__global__ void k_softmax(const float* __restrict__ corr, bf16_t* __restrict__ attn) {
    const int row = blockIdx.x;
    const int tid = threadIdx.x;
    __shared__ float sh[256];
    const float* p = corr + (size_t)row * HW;
    float v[4];
    float mx = -3.0e38f;
#pragma unroll
    for (int i = 0; i < 4; ++i) { v[i] = p[tid + i * 256] * 20.0f; mx = fmaxf(mx, v[i]); }
    sh[tid] = mx;
    __syncthreads();
    for (int s = 128; s > 0; s >>= 1) { if (tid < s) sh[tid] = fmaxf(sh[tid], sh[tid + s]); __syncthreads(); }
    mx = sh[0];
    __syncthreads();
    float e[4], sum = 0.f;
#pragma unroll
    for (int i = 0; i < 4; ++i) { e[i] = __expf(v[i] - mx); sum += e[i]; }
    sh[tid] = sum;
    __syncthreads();
    for (int s = 128; s > 0; s >>= 1) { if (tid < s) sh[tid] += sh[tid + s]; __syncthreads(); }
    const float inv = 1.0f / sh[0];
#pragma unroll
    for (int i = 0; i < 4; ++i)
        attn[(size_t)row * HW + tid + i * 256] = (bf16_t)(e[i] * inv);
}

// ---------------------------------------------------------------------------
// K4: att_fq[b][q][c] = sum_s attn[b][q][s] * f_s[b][c][s].  A=attn rows,
//     B rows = fsb[b][c][*] (s contiguous).  grid=(c_tiles=16, q_tiles=32, B)
// ---------------------------------------------------------------------------
__global__ void k_attgemm(const bf16_t* __restrict__ attn, const bf16_t* __restrict__ fsb,
                          float* __restrict__ attws) {
    const int c0 = blockIdx.x * 32, q0 = blockIdx.y * 32, b = blockIdx.z;
    const int lane = threadIdx.x, m = lane & 15, half = lane >> 4;
    const bf16_t* A = attn + (size_t)b * HW * HW;
    const bf16_t* a0r = A + (size_t)(q0 + m) * HW;
    const bf16_t* a1r = A + (size_t)(q0 + 16 + m) * HW;
    const bf16_t* Bm = fsb + (size_t)b * CC * HW;
    const bf16_t* b0r = Bm + (size_t)(c0 + m) * HW;
    const bf16_t* b1r = Bm + (size_t)(c0 + 16 + m) * HW;
    v8f c00 = v8f_zero(), c01 = v8f_zero(), c10 = v8f_zero(), c11 = v8f_zero();
#pragma unroll 4
    for (int kc = 0; kc < 32; ++kc) {
        const int k0 = kc * 32;
        v16bf a0 = load_a16(a0r, k0, half);
        v16bf a1 = load_a16(a1r, k0, half);
        v16bf bm0 = load_b16(b0r, k0, half);
        v16bf bm1 = load_b16(b1r, k0, half);
        c00 = wmma_bf16(a0, bm0, c00);
        c01 = wmma_bf16(a0, bm1, c01);
        c10 = wmma_bf16(a1, bm0, c10);
        c11 = wmma_bf16(a1, bm1, c11);
    }
    float* out = attws + (size_t)b * HW * CC;
#pragma unroll
    for (int r = 0; r < 8; ++r) {
        int row0 = q0 + r + 8 * half, row1 = row0 + 16;
        out[(size_t)row0 * CC + c0 + m]      = c00[r];
        out[(size_t)row0 * CC + c0 + 16 + m] = c01[r];
        out[(size_t)row1 * CC + c0 + m]      = c10[r];
        out[(size_t)row1 * CC + c0 + 16 + m] = c11[r];
    }
}

// ---------------------------------------------------------------------------
// K5: fq = l2n(f_q) + l2n(att_fq)*0.5 (norm over c); also emit raw att_fq in
//     [B,C,h,w]. One 128-thread block per (b,q).
// ---------------------------------------------------------------------------
__global__ void k_finalize(const float* __restrict__ attws, const float* __restrict__ f_q,
                           float* __restrict__ out_fq, float* __restrict__ out_att) {
    const int q = blockIdx.x, b = blockIdx.y, tid = threadIdx.x;  // 128 threads
    __shared__ float sA[128], sQ[128];
    const float* arow = attws + ((size_t)b * HW + q) * CC;
    float av[4], qv[4], ssa = 0.f, ssq = 0.f;
#pragma unroll
    for (int i = 0; i < 4; ++i) {
        int c = tid + i * 128;
        av[i] = arow[c];
        qv[i] = f_q[((size_t)b * CC + c) * HW + q];
        ssa += av[i] * av[i];
        ssq += qv[i] * qv[i];
    }
    sA[tid] = ssa; sQ[tid] = ssq;
    __syncthreads();
    for (int s = 64; s > 0; s >>= 1) {
        if (tid < s) { sA[tid] += sA[tid + s]; sQ[tid] += sQ[tid + s]; }
        __syncthreads();
    }
    const float ia = 1.0f / fmaxf(sqrtf(sA[0]), 1e-12f);
    const float iq = 1.0f / fmaxf(sqrtf(sQ[0]), 1e-12f);
#pragma unroll
    for (int i = 0; i < 4; ++i) {
        int c = tid + i * 128;
        size_t idx = ((size_t)b * CC + c) * HW + q;
        out_fq[idx]  = qv[i] * iq + av[i] * ia * 0.5f;
        out_att[idx] = av[i];
    }
}

// ---------------------------------------------------------------------------
// K6: conv1 as 9-tap implicit GEMM (pad=2, dil=2): M=1024 pos, N=256, K=1024.
//     comp NHWC bf16 [b][p][1024]; w1t [tap][co][ci]. ReLU+bias -> x1 bf16.
// ---------------------------------------------------------------------------
__global__ void k_conv1(const bf16_t* __restrict__ comp, const bf16_t* __restrict__ w1t,
                        const float* __restrict__ bias, bf16_t* __restrict__ x1) {
    const int n0 = blockIdx.x * 32, q0 = blockIdx.y * 32, b = blockIdx.z;
    const int lane = threadIdx.x, m = lane & 15, half = lane >> 4;
    v8f c00 = v8f_zero(), c01 = v8f_zero(), c10 = v8f_zero(), c11 = v8f_zero();
    const int r0 = q0 + m, r1 = q0 + 16 + m;
    const int y0 = r0 >> 5, x0 = r0 & 31, y1 = r1 >> 5, x1c = r1 & 31;
    const bf16_t* cb = comp + (size_t)b * HW * 1024;
    for (int t = 0; t < 9; ++t) {
        const int dy = (t / 3) * 2 - 2, dx = (t % 3) * 2 - 2;
        const int sy0 = y0 + dy, sx0 = x0 + dx, sy1 = y1 + dy, sx1 = x1c + dx;
        const bool v0 = ((unsigned)sy0 < 32u) && ((unsigned)sx0 < 32u);
        const bool v1 = ((unsigned)sy1 < 32u) && ((unsigned)sx1 < 32u);
        const bf16_t* a0r = cb + (size_t)(v0 ? (sy0 * 32 + sx0) : 0) * 1024;
        const bf16_t* a1r = cb + (size_t)(v1 ? (sy1 * 32 + sx1) : 0) * 1024;
        const bf16_t* b0r = w1t + (size_t)(t * 256 + n0 + m) * 1024;
        const bf16_t* b1r = w1t + (size_t)(t * 256 + n0 + 16 + m) * 1024;
#pragma unroll 4
        for (int kc = 0; kc < 32; ++kc) {
            const int k0 = kc * 32;
            v16bf a0 = v0 ? load_a16(a0r, k0, half) : v16bf_zero();
            v16bf a1 = v1 ? load_a16(a1r, k0, half) : v16bf_zero();
            v16bf bm0 = load_b16(b0r, k0, half);
            v16bf bm1 = load_b16(b1r, k0, half);
            c00 = wmma_bf16(a0, bm0, c00);
            c01 = wmma_bf16(a0, bm1, c01);
            c10 = wmma_bf16(a1, bm0, c10);
            c11 = wmma_bf16(a1, bm1, c11);
        }
    }
    bf16_t* outp = x1 + (size_t)b * HW * 256;
    const int cA = n0 + m, cB = n0 + 16 + m;
    const float bA = bias[cA], bB = bias[cB];
#pragma unroll
    for (int r = 0; r < 8; ++r) {
        int row0 = q0 + r + 8 * half, row1 = row0 + 16;
        outp[(size_t)row0 * 256 + cA] = (bf16_t)fmaxf(c00[r] + bA, 0.f);
        outp[(size_t)row0 * 256 + cB] = (bf16_t)fmaxf(c01[r] + bB, 0.f);
        outp[(size_t)row1 * 256 + cA] = (bf16_t)fmaxf(c10[r] + bA, 0.f);
        outp[(size_t)row1 * 256 + cB] = (bf16_t)fmaxf(c11[r] + bB, 0.f);
    }
}

// ---------------------------------------------------------------------------
// K7: conv2 (pad=0): M=900 (30x30), N=256, K=256, 9 taps over x1 (32x32 NHWC).
// ---------------------------------------------------------------------------
__global__ void k_conv2(const bf16_t* __restrict__ x1, const bf16_t* __restrict__ w2t,
                        const float* __restrict__ bias, float* __restrict__ x2) {
    const int n0 = blockIdx.x * 32, q0 = blockIdx.y * 32, b = blockIdx.z;
    const int lane = threadIdx.x, m = lane & 15, half = lane >> 4;
    v8f c00 = v8f_zero(), c01 = v8f_zero(), c10 = v8f_zero(), c11 = v8f_zero();
    const int r0 = q0 + m, r1 = q0 + 16 + m;
    const bool v0 = r0 < 900, v1 = r1 < 900;
    const int y0 = r0 / 30, x0 = r0 % 30, y1 = r1 / 30, x1p = r1 % 30;
    const bf16_t* xb = x1 + (size_t)b * HW * 256;
    for (int t = 0; t < 9; ++t) {
        const int dy = t / 3, dx = t % 3;
        const bf16_t* a0r = xb + (size_t)(v0 ? ((y0 + dy) * 32 + x0 + dx) : 0) * 256;
        const bf16_t* a1r = xb + (size_t)(v1 ? ((y1 + dy) * 32 + x1p + dx) : 0) * 256;
        const bf16_t* b0r = w2t + (size_t)(t * 256 + n0 + m) * 256;
        const bf16_t* b1r = w2t + (size_t)(t * 256 + n0 + 16 + m) * 256;
#pragma unroll 4
        for (int kc = 0; kc < 8; ++kc) {
            const int k0 = kc * 32;
            v16bf a0 = v0 ? load_a16(a0r, k0, half) : v16bf_zero();
            v16bf a1 = v1 ? load_a16(a1r, k0, half) : v16bf_zero();
            v16bf bm0 = load_b16(b0r, k0, half);
            v16bf bm1 = load_b16(b1r, k0, half);
            c00 = wmma_bf16(a0, bm0, c00);
            c01 = wmma_bf16(a0, bm1, c01);
            c10 = wmma_bf16(a1, bm0, c10);
            c11 = wmma_bf16(a1, bm1, c11);
        }
    }
    float* outp = x2 + (size_t)b * 900 * 256;
    const int cA = n0 + m, cB = n0 + 16 + m;
    const float bA = bias[cA], bB = bias[cB];
#pragma unroll
    for (int r = 0; r < 8; ++r) {
        int row0 = q0 + r + 8 * half, row1 = row0 + 16;
        if (row0 < 900) {
            outp[(size_t)row0 * 256 + cA] = c00[r] + bA;
            outp[(size_t)row0 * 256 + cB] = c01[r] + bB;
        }
        if (row1 < 900) {
            outp[(size_t)row1 * 256 + cA] = c10[r] + bA;
            outp[(size_t)row1 * 256 + cB] = c11[r] + bB;
        }
    }
}

// K8: 3x3/3 maxpool on NHWC [b][30][30][256] -> [b][10][10][256]
__global__ void k_maxpool(const float* __restrict__ x2, float* __restrict__ pooled) {
    int idx = blockIdx.x * 256 + threadIdx.x;
    if (idx >= 4 * 100 * 256) return;
    int c = idx & 255;
    int rest = idx >> 8;
    int px = rest % 10, py = (rest / 10) % 10, b = rest / 100;
    float mv = -3.0e38f;
#pragma unroll
    for (int i = 0; i < 3; ++i)
#pragma unroll
        for (int j = 0; j < 3; ++j) {
            int p = (3 * py + i) * 30 + 3 * px + j;
            mv = fmaxf(mv, x2[((size_t)b * 900 + p) * 256 + c]);
        }
    pooled[idx] = mv;
}

// K9: conv3 (1 out-ch, 3x3) on [b][10][10][256] -> mean over 8x8 -> weight[b]
__global__ void k_conv3(const float* __restrict__ pooled, const float* __restrict__ w3,
                        const float* __restrict__ b3, float* __restrict__ wout) {
    const int b = blockIdx.x, tid = threadIdx.x;  // 64 threads
    __shared__ float sh[64];
    const int oy = tid >> 3, ox = tid & 7;
    float acc = b3[0];
    for (int dy = 0; dy < 3; ++dy)
        for (int dx = 0; dx < 3; ++dx) {
            const float* prow = pooled + ((size_t)b * 100 + (oy + dy) * 10 + ox + dx) * 256;
            for (int c = 0; c < 256; ++c)
                acc += prow[c] * w3[(c * 3 + dy) * 3 + dx];
        }
    sh[tid] = acc;
    __syncthreads();
    for (int s = 32; s > 0; s >>= 1) { if (tid < s) sh[tid] += sh[tid + s]; __syncthreads(); }
    if (tid == 0) wout[b] = sh[0] / 64.0f;
}

// ---------------------------------------------------------------------------
extern "C" void kernel_launch(void* const* d_in, const int* in_sizes, int n_in,
                              void* d_out, int out_size, void* d_ws, size_t ws_size,
                              hipStream_t stream) {
    const float* fq_feats = (const float*)d_in[0];
    const float* fs_feats = (const float*)d_in[1];
    const float* f_q = (const float*)d_in[2];
    const float* f_s = (const float*)d_in[3];
    const float* w1 = (const float*)d_in[4];
    const float* b1 = (const float*)d_in[5];
    const float* w2 = (const float*)d_in[6];
    const float* b2 = (const float*)d_in[7];
    const float* w3 = (const float*)d_in[8];
    const float* b3 = (const float*)d_in[9];

    char* ws = (char*)d_ws;
    bf16_t* qn   = (bf16_t*)(ws + QN_OFF);
    bf16_t* sn   = (bf16_t*)(ws + SN_OFF);
    float*  corr = (float*) (ws + CORR_OFF);
    bf16_t* fsb  = (bf16_t*)(ws + FSB_OFF);
    bf16_t* attn = (bf16_t*)(ws + ATTN_OFF);
    float*  attws= (float*) (ws + ATTWS_OFF);
    bf16_t* comp = (bf16_t*)(ws + COMP_OFF);
    bf16_t* w1t  = (bf16_t*)(ws + W1T_OFF);
    bf16_t* x1   = (bf16_t*)(ws + X1_OFF);
    bf16_t* w2t  = (bf16_t*)(ws + W2T_OFF);
    float*  x2   = (float*) (ws + X2_OFF);
    float*  pooled = (float*)(ws + POOL_OFF);

    float* out_fq  = (float*)d_out;
    float* out_att = out_fq + (size_t)BB * CC * HW;       // 2,097,152
    float* out_w   = out_fq + (size_t)2 * BB * CC * HW;   // 4,194,304

    // Phase 1: operand prep that must precede k_corr (no overlay conflicts)
    k_transpose_norm<<<dim3(LVL * BB, 64), 256, 0, stream>>>(fq_feats, qn, 512, 0, 1);
    k_transpose_norm<<<dim3(LVL * BB, 64), 256, 0, stream>>>(fs_feats, sn, 512, 0, 1);
    k_cvt_bf16<<<8192, 256, 0, stream>>>(f_s, fsb, BB * CC * HW);

    // Phase 2: correlation GEMM (reads qn/sn)
    k_corr<<<dim3(32, 32, BB), 32, 0, stream>>>(qn, sn, corr);

    // Phase 3: softmax (attn overlays qn region — qn is dead now)
    k_softmax<<<BB * HW, 256, 0, stream>>>(corr, attn);

    // Conv-branch prep (overlays sn region — sn is dead now)
    k_transpose_norm<<<dim3(BB, 64), 256, 0, stream>>>(f_q, comp, 1024, 0, 0);
    k_transpose_norm<<<dim3(BB, 64), 256, 0, stream>>>(f_s, comp, 1024, 512, 0);
    k_wxform<<<9216, 256, 0, stream>>>(w1, w1t, 256, 1024);
    k_wxform<<<2304, 256, 0, stream>>>(w2, w2t, 256, 256);

    // Phase 4: attention aggregation GEMM + finalize
    k_attgemm<<<dim3(16, 32, BB), 32, 0, stream>>>(attn, fsb, attws);
    k_finalize<<<dim3(HW, BB), 128, 0, stream>>>(attws, f_q, out_fq, out_att);

    // Phase 5: conv branch
    k_conv1<<<dim3(8, 32, BB), 32, 0, stream>>>(comp, w1t, b1, x1);
    k_conv2<<<dim3(8, 29, BB), 32, 0, stream>>>(x1, w2t, b2, x2);
    k_maxpool<<<400, 256, 0, stream>>>(x2, pooled);
    k_conv3<<<BB, 64, 0, stream>>>(pooled, w3, b3, out_w);
}
